// MultiHeadAttention_14894946583119
// MI455X (gfx1250) — matrix operand order repaired and checked by reading, not compiled, not run on hardware
//
#include <hip/hip_runtime.h>

// ---------------------------------------------------------------------------
// MHA forward for B=4, T=2048, C=1024, H=16, D=64 on gfx1250 (MI455X).
// f16 WMMA (v_wmma_f32_16x16x32_f16) with f32 accumulation everywhere,
// flash-attention online softmax, V stored pre-transposed, K/V tiles staged
// in LDS once per workgroup (shared by 4 waves), async global->LDS copies
// (GLOBAL_LOAD_ASYNC_TO_LDS_B128 / ASYNCcnt) when the toolchain exposes them.
// ---------------------------------------------------------------------------

typedef __attribute__((ext_vector_type(16))) _Float16 v16h;
typedef __attribute__((ext_vector_type(8)))  float    v8f;
typedef __attribute__((ext_vector_type(4)))  int      v4i;

#define NEG_INF (-3.0e38f)

#define WMMA_F16(a, b, c) \
  __builtin_amdgcn_wmma_f32_16x16x32_f16(false, (a), false, (b), (short)0, (c), false, false)

#if defined(__HIP_DEVICE_COMPILE__) && defined(__gfx1250__) && \
    __has_builtin(__builtin_amdgcn_global_load_async_to_lds_b128)
#define HAVE_ASYNC_LDS 1
#else
#define HAVE_ASYNC_LDS 0
#endif

union FragU { v16h v; uint4 q[2]; };
union Pack8 { uint4 q; _Float16 h[8]; };
union Pack4 { uint2 u; _Float16 h[4]; };

#if HAVE_ASYNC_LDS
// 16-byte async copy global -> LDS (tracked by ASYNCcnt, bypasses VGPRs).
// Builtin signature (from hipcc diagnostic): (v4i AS1*, v4i AS3*, imm, imm).
__device__ __forceinline__ void async_cp16(const _Float16* g, _Float16* l) {
  __builtin_amdgcn_global_load_async_to_lds_b128(
      (__attribute__((address_space(1))) v4i*)(g),
      (__attribute__((address_space(3))) v4i*)(l),
      0, 0);
}
__device__ __forceinline__ void wait_async0() {
#if __has_builtin(__builtin_amdgcn_s_wait_asynccnt)
  __builtin_amdgcn_s_wait_asynccnt(0);
#else
  asm volatile("s_wait_asynccnt 0x0" ::: "memory");
#endif
}
#endif

// Load one WMMA A/B fragment: two 16-byte chunks (K={kf..kf+7},{kf+16..kf+23}).
__device__ __forceinline__ v16h load_frag(const _Float16* p0, const _Float16* p1) {
  FragU f;
  f.q[0] = *(const uint4*)p0;
  f.q[1] = *(const uint4*)p1;
  return f.v;
}

__device__ __forceinline__ v8f zero8() {
  v8f z = {0.f, 0.f, 0.f, 0.f, 0.f, 0.f, 0.f, 0.f};
  return z;
}

// Butterfly reductions across a 16-lane group (wave32: two groups per wave).
__device__ __forceinline__ float redmax16(float x) {
  x = fmaxf(x, __shfl_xor(x, 1, 32));
  x = fmaxf(x, __shfl_xor(x, 2, 32));
  x = fmaxf(x, __shfl_xor(x, 4, 32));
  x = fmaxf(x, __shfl_xor(x, 8, 32));
  return x;
}
__device__ __forceinline__ float redsum16(float x) {
  x += __shfl_xor(x, 1, 32);
  x += __shfl_xor(x, 2, 32);
  x += __shfl_xor(x, 4, 32);
  x += __shfl_xor(x, 8, 32);
  return x;
}

// ---------------------------------------------------------------------------
// f32 -> f16 conversion, 4 elements per thread.
// ---------------------------------------------------------------------------
__global__ void cvt_f32_f16(const float* __restrict__ in, _Float16* __restrict__ out, int n4) {
  int i = blockIdx.x * 256 + threadIdx.x;
  if (i < n4) {
    float4 v = ((const float4*)in)[i];
    Pack4 p;
    p.h[0] = (_Float16)v.x;
    p.h[1] = (_Float16)v.y;
    p.h[2] = (_Float16)v.z;
    p.h[3] = (_Float16)v.w;
    ((uint2*)out)[i] = p.u;
  }
}

// ---------------------------------------------------------------------------
// 64x64-tile WMMA GEMM, 128 threads (4 waves), K-step 32.
// MODE 0: QKV projection -> scatter Q,K as [BH][T][D] f16; V transposed
//         as [BH][D][T] f16 (so attention never needs a transpose).
// MODE 1: output projection -> f32 to d_out, + bias.
// ---------------------------------------------------------------------------
template <int MODE>
__global__ __launch_bounds__(128)
void gemm64x64(const _Float16* __restrict__ Ag, const _Float16* __restrict__ Bg,
               const float* __restrict__ bias, float* __restrict__ Cf32,
               _Float16* __restrict__ q16, _Float16* __restrict__ k16,
               _Float16* __restrict__ vT, int K, int ldb) {
  __shared__ _Float16 As[64][40];   // [m][k], +8 pad
  __shared__ _Float16 Bs[64][40];   // [n][k] (transposed in), +8 pad

  const int tid   = threadIdx.x;
  const int mBase = blockIdx.x * 64;
  const int nBase = blockIdx.y * 64;
  const int wv    = tid >> 5;
  const int lane  = tid & 31;
  const int grp   = lane >> 4;          // 0: K 0..7/16..23, 1: K 8..15/24..31
  const int lm    = lane & 15;
  const int kf    = grp * 8;
  const int mw    = (wv >> 1) * 32;     // wave's 32x32 sub-tile
  const int nw    = (wv & 1) * 32;

  v8f acc[2][2];
#pragma unroll
  for (int i = 0; i < 2; ++i)
#pragma unroll
    for (int j = 0; j < 2; ++j) acc[i][j] = zero8();

  for (int k0 = 0; k0 < K; k0 += 32) {
    __syncthreads();
    // Stage A tile 64x32 f16: 256 16B chunks / 128 threads.
#if HAVE_ASYNC_LDS
#pragma unroll
    for (int i = 0; i < 2; ++i) {
      int c = tid + i * 128;
      int row = c >> 2, cc = (c & 3) * 8;
      async_cp16(Ag + (size_t)(mBase + row) * K + k0 + cc, &As[row][cc]);
    }
#else
#pragma unroll
    for (int i = 0; i < 2; ++i) {
      int c = tid + i * 128;
      int row = c >> 2, cc = (c & 3) * 8;
      *(uint4*)&As[row][cc] = *(const uint4*)(Ag + (size_t)(mBase + row) * K + k0 + cc);
    }
#endif
    // Stage B tile 32x64 f16, transposed into Bs[n][k].
#pragma unroll
    for (int i = 0; i < 2; ++i) {
      int c = tid + i * 128;
      int kk = c >> 3, nc = (c & 7) * 8;
      Pack8 p;
      p.q = *(const uint4*)(Bg + (size_t)(k0 + kk) * ldb + nBase + nc);
#pragma unroll
      for (int j = 0; j < 8; ++j) Bs[nc + j][kk] = p.h[j];
    }
#if HAVE_ASYNC_LDS
    wait_async0();                       // barriers do not wait on ASYNCcnt
#endif
    __syncthreads();

    // Prefetch next K-step tiles into cache while this one computes.
    if (k0 + 32 < K) {
      __builtin_prefetch(Ag + (size_t)(mBase + (tid >> 1)) * K + k0 + 32 + (tid & 1) * 16, 0, 1);
      __builtin_prefetch(Bg + (size_t)(k0 + 32 + (tid >> 2)) * ldb + nBase + (tid & 3) * 16, 0, 1);
    }

    v16h a0 = load_frag(&As[mw + lm][kf],      &As[mw + lm][kf + 16]);
    v16h a1 = load_frag(&As[mw + 16 + lm][kf], &As[mw + 16 + lm][kf + 16]);
    v16h b0 = load_frag(&Bs[nw + lm][kf],      &Bs[nw + lm][kf + 16]);
    v16h b1 = load_frag(&Bs[nw + 16 + lm][kf], &Bs[nw + 16 + lm][kf + 16]);

    acc[0][0] = WMMA_F16(a0, b0, acc[0][0]);
    acc[0][1] = WMMA_F16(a0, b1, acc[0][1]);
    acc[1][0] = WMMA_F16(a1, b0, acc[1][0]);
    acc[1][1] = WMMA_F16(a1, b1, acc[1][1]);
  }

  // Epilogue. C layout: VGPR r -> M = r + grp*8 ; N = lm (per 16x16 subtile).
#pragma unroll
  for (int ms = 0; ms < 2; ++ms) {
    const int tG = mBase + mw + ms * 16 + grp * 8;  // global row of r=0
    const int bb = tG >> 11;                         // row / T
    const int tt = tG & 2047;                        // row % T
#pragma unroll
    for (int ns = 0; ns < 2; ++ns) {
      const int nG = nBase + nw + ns * 16 + lm;
      const float bv = bias[nG];
      if (MODE == 0) {
        const int which = nG >> 10;        // 0=q 1=k 2=v
        const int rem = nG & 1023;
        const int hh = rem >> 6, dd = rem & 63;
        const int bhh = bb * 16 + hh;
        if (which == 2) {
          // V transposed: vT[bh][d][t]; 8 consecutive t per lane -> one b128.
          Pack8 p;
#pragma unroll
          for (int r = 0; r < 8; ++r) p.h[r] = (_Float16)(acc[ms][ns][r] + bv);
          *(uint4*)(vT + ((size_t)bhh * 64 + dd) * 2048 + tt) = p.q;
        } else {
          _Float16* dst = (which == 0 ? q16 : k16) + ((size_t)bhh * 2048 + tt) * 64 + dd;
#pragma unroll
          for (int r = 0; r < 8; ++r) dst[(size_t)r * 64] = (_Float16)(acc[ms][ns][r] + bv);
        }
      } else {
        float* dst = Cf32 + (size_t)tG * 1024 + nG;
#pragma unroll
        for (int r = 0; r < 8; ++r) dst[(size_t)r * 1024] = acc[ms][ns][r] + bv;
      }
    }
  }
}

// ---------------------------------------------------------------------------
// Flash attention: one block per (b,h, 64 query rows); each wave owns 16 rows.
// K and V(^T) tiles are staged once per block in LDS (shared by all 4 waves,
// async copies when available); P goes through a per-wave LDS buffer for the
// C->A fragment relayout. All waves share the same kt trip count.
// ---------------------------------------------------------------------------
__global__ __launch_bounds__(128)
void attn_fa(const _Float16* __restrict__ q16, const _Float16* __restrict__ k16,
             const _Float16* __restrict__ vT, _Float16* __restrict__ o16) {
  __shared__ _Float16 Ks[64][64];      // K tile   [kv][d]   (B-frag layout)
  __shared__ _Float16 Vs[64][64];      // V^T tile [d][kv]   (B-frag layout)
  __shared__ _Float16 Ps[4][16][72];   // per-wave P tile [q=16][kv=64], +8 pad

  const int tid  = threadIdx.x;
  const int wv   = tid >> 5;
  const int lane = tid & 31;
  const int grp  = lane >> 4;
  const int lm   = lane & 15;
  const int kf   = grp * 8;
  const int qt   = blockIdx.x;           // 64-row query tile
  const int bh   = blockIdx.y;           // b*16 + h
  const int b    = bh >> 4, h = bh & 15;
  const int qwave = qt * 64 + wv * 16;   // first query row of this wave
  const int qrow0 = qwave + grp * 8;     // query row of C-element r=0

  const _Float16* kp = k16 + (size_t)bh * 2048 * 64;  // [t][d]
  const _Float16* vp = vT  + (size_t)bh * 64 * 2048;  // [d][t]

  // Q fragments (held for the whole kernel).
  const _Float16* qr = q16 + ((size_t)bh * 2048 + qwave + lm) * 64;
  const v16h qa0 = load_frag(qr + kf, qr + kf + 16);
  const v16h qa1 = load_frag(qr + 32 + kf, qr + 32 + kf + 16);

  float mi[8], li[8];
  v8f oc[4];
#pragma unroll
  for (int r = 0; r < 8; ++r) { mi[r] = NEG_INF; li[r] = 0.f; }
#pragma unroll
  for (int dt = 0; dt < 4; ++dt) oc[dt] = zero8();

  for (int kt = 0; kt <= qt; ++kt) {
    const int kb = kt * 64;

    // Stage K tile (8KB) and V^T tile (8KB) into LDS, shared by all waves.
    __syncthreads();   // previous iteration's tile reads are done
#if HAVE_ASYNC_LDS
#pragma unroll
    for (int i = 0; i < 4; ++i) {
      int c = tid + i * 128;             // 512 chunks of 16B
      int row = c >> 3, cc = (c & 7) * 8;
      async_cp16(kp + (size_t)(kb + row) * 64 + cc, &Ks[row][cc]);
      async_cp16(vp + (size_t)row * 2048 + kb + cc, &Vs[row][cc]);
    }
    wait_async0();
#else
#pragma unroll
    for (int i = 0; i < 4; ++i) {
      int c = tid + i * 128;
      int row = c >> 3, cc = (c & 7) * 8;
      *(uint4*)&Ks[row][cc] = *(const uint4*)(kp + (size_t)(kb + row) * 64 + cc);
      *(uint4*)&Vs[row][cc] = *(const uint4*)(vp + (size_t)row * 2048 + kb + cc);
    }
#endif
    __syncthreads();

    // Prefetch next kv-tile while computing this one.
    if (kt < qt) {
      __builtin_prefetch(kp + (size_t)(kb + 64 + (tid >> 1)) * 64 + (tid & 1) * 32, 0, 1);
      __builtin_prefetch(vp + (size_t)(tid >> 1) * 2048 + kb + 64 + (tid & 1) * 32, 0, 1);
    }

    // S = (Q K^T) * scale, 16x64 per wave (fragments from LDS).
    v8f sf[4];
#pragma unroll
    for (int nt = 0; nt < 4; ++nt) {
      const _Float16* kr = &Ks[nt * 16 + lm][0];
      v16h kb0 = load_frag(kr + kf, kr + kf + 16);
      v16h kb1 = load_frag(kr + 32 + kf, kr + 32 + kf + 16);
      v8f s = zero8();
      s = WMMA_F16(qa0, kb0, s);
      s = WMMA_F16(qa1, kb1, s);
      sf[nt] = s;
    }

    // scale + causal mask (diagonal tile only).
#pragma unroll
    for (int nt = 0; nt < 4; ++nt) {
      const int kcol = kb + nt * 16 + lm;
#pragma unroll
      for (int r = 0; r < 8; ++r) {
        float v = sf[nt][r] * 0.125f;  // D^-0.5
        if (kt == qt && kcol > qrow0 + r) v = NEG_INF;
        sf[nt][r] = v;
      }
    }

    // Online softmax: row max / rescale / exp / row sum.
    float mx[8], al[8], rs[8];
#pragma unroll
    for (int r = 0; r < 8; ++r) {
      float m = fmaxf(fmaxf(sf[0][r], sf[1][r]), fmaxf(sf[2][r], sf[3][r]));
      mx[r] = redmax16(m);
    }
#pragma unroll
    for (int r = 0; r < 8; ++r) {
      float mn = fmaxf(mi[r], mx[r]);
      al[r] = __expf(mi[r] - mn);
      mi[r] = mn;
      rs[r] = 0.f;
    }
#pragma unroll
    for (int nt = 0; nt < 4; ++nt)
#pragma unroll
      for (int r = 0; r < 8; ++r) {
        float p = __expf(sf[nt][r] - mi[r]);
        sf[nt][r] = p;
        rs[r] += p;
      }
#pragma unroll
    for (int r = 0; r < 8; ++r) {
      li[r] = li[r] * al[r] + redsum16(rs[r]);
    }
#pragma unroll
    for (int dt = 0; dt < 4; ++dt)
#pragma unroll
      for (int r = 0; r < 8; ++r) oc[dt][r] *= al[r];

    // P: C-layout -> LDS -> A-layout (same-wave DS ops are in-order).
#pragma unroll
    for (int nt = 0; nt < 4; ++nt)
#pragma unroll
      for (int r = 0; r < 8; ++r)
        Ps[wv][grp * 8 + r][nt * 16 + lm] = (_Float16)sf[nt][r];

    const _Float16* pr = &Ps[wv][lm][0];
    v16h pa0 = load_frag(pr + kf, pr + kf + 16);
    v16h pa1 = load_frag(pr + 32 + kf, pr + 32 + kf + 16);

    // O += P V  (V^T tile from LDS).
#pragma unroll
    for (int dt = 0; dt < 4; ++dt) {
      const _Float16* vr = &Vs[dt * 16 + lm][0];
      v16h vb0 = load_frag(vr + kf, vr + kf + 16);
      v16h vb1 = load_frag(vr + 32 + kf, vr + 32 + kf + 16);
      oc[dt] = WMMA_F16(pa0, vb0, oc[dt]);
      oc[dt] = WMMA_F16(pa1, vb1, oc[dt]);
    }
  }

  // Normalize and store attention output as [B,T,H,D] = [B*T, C] f16.
  float inv[8];
#pragma unroll
  for (int r = 0; r < 8; ++r) inv[r] = 1.0f / li[r];
#pragma unroll
  for (int dt = 0; dt < 4; ++dt)
#pragma unroll
    for (int r = 0; r < 8; ++r) {
      const size_t row = (size_t)b * 2048 + qwave + grp * 8 + r;
      o16[row * 1024 + h * 64 + dt * 16 + lm] = (_Float16)(oc[dt][r] * inv[r]);
    }
}

// ---------------------------------------------------------------------------
// Host launch
// ---------------------------------------------------------------------------
extern "C" void kernel_launch(void* const* d_in, const int* in_sizes, int n_in,
                              void* d_out, int out_size, void* d_ws, size_t ws_size,
                              hipStream_t stream) {
  (void)in_sizes; (void)n_in; (void)out_size; (void)ws_size;
  const float* x     = (const float*)d_in[0];
  const float* w_qkv = (const float*)d_in[1];
  const float* b_qkv = (const float*)d_in[2];
  const float* w_out = (const float*)d_in[3];
  const float* b_out = (const float*)d_in[4];
  float* out = (float*)d_out;

  char* ws = (char*)d_ws;
  _Float16* x16  = (_Float16*)(ws);                       // 8192*1024   f16
  _Float16* wq16 = (_Float16*)(ws + 16777216);            // 1024*3072   f16
  _Float16* wo16 = (_Float16*)(ws + 23068672);            // 1024*1024   f16
  _Float16* q16  = (_Float16*)(ws + 25165824);            // [64][2048][64]
  _Float16* k16  = (_Float16*)(ws + 41943040);            // [64][2048][64]
  _Float16* vT   = (_Float16*)(ws + 58720256);            // [64][64][2048]
  _Float16* o16  = (_Float16*)(ws + 75497472);            // 8192*1024   f16

  cvt_f32_f16<<<8192, 256, 0, stream>>>(x,     x16,  2097152);
  cvt_f32_f16<<<3072, 256, 0, stream>>>(w_qkv, wq16, 786432);
  cvt_f32_f16<<<1024, 256, 0, stream>>>(w_out, wo16, 262144);

  gemm64x64<0><<<dim3(128, 48), 128, 0, stream>>>(x16, wq16, b_qkv, nullptr,
                                                  q16, k16, vT, 1024, 3072);

  attn_fa<<<dim3(32, 64), 128, 0, stream>>>(q16, k16, vT, o16);

  gemm64x64<1><<<dim3(128, 16), 128, 0, stream>>>(o16, wo16, b_out, out,
                                                  nullptr, nullptr, nullptr, 1024, 1024);
}
